// MultiHeadGraphAttention_85770496901458
// MI455X (gfx1250) — matrix operand order repaired and checked by reading, not compiled, not run on hardware
//
#include <hip/hip_runtime.h>

#define UNITS 256
#define NUM_HEADS 8
#define HEAD_DIM 32
#define BB 8
#define NN 2048
#define NWORDS (NN / 32)
#define SCALE 0.1767766952966369f   // 1/sqrt(32)

typedef _Float16 h8  __attribute__((ext_vector_type(8)));
typedef _Float16 h16 __attribute__((ext_vector_type(16)));
typedef float    f8  __attribute__((ext_vector_type(8)));

// Build a 16-half A/B fragment from two contiguous 8-half (16B) runs.
__device__ __forceinline__ h16 ld_h16(const _Float16* __restrict__ p0,
                                      const _Float16* __restrict__ p1) {
  h8 lo = *(const h8*)p0;
  h8 hi = *(const h8*)p1;
  h16 r;
#pragma unroll
  for (int i = 0; i < 8; ++i) { r[i] = lo[i]; r[i + 8] = hi[i]; }
  return r;
}

__device__ __forceinline__ void cvt8(h16& r, int base, float4 u, float4 v) {
  r[base + 0] = (_Float16)u.x; r[base + 1] = (_Float16)u.y;
  r[base + 2] = (_Float16)u.z; r[base + 3] = (_Float16)u.w;
  r[base + 4] = (_Float16)v.x; r[base + 5] = (_Float16)v.y;
  r[base + 6] = (_Float16)v.z; r[base + 7] = (_Float16)v.w;
}

// ---------------------------------------------------------------------------
// Kernel 0: pack adjacency int32 -> bitmask.  One wave32 ballot = one u32 word
// covering 32 keys.  Cuts attention-phase mask traffic 32x (and 4B -> bit).
// ---------------------------------------------------------------------------
__global__ __launch_bounds__(256) void pack_adj_kernel(
    const int* __restrict__ adj, unsigned* __restrict__ packed) {
  const int row  = blockIdx.x;               // b*N + q  (0..16383)
  const int wv   = threadIdx.x >> 5;
  const int lane = threadIdx.x & 31;
  const int* arow = adj + (size_t)row * NN;
  unsigned* prow  = packed + (size_t)row * NWORDS;
  for (int w = wv; w < NWORDS; w += 8) {
    const int v = arow[w * 32 + lane];
    const unsigned long long m = __ballot(v > 0);
    if (lane == 0) prow[w] = (unsigned)m;
  }
}

// ---------------------------------------------------------------------------
// Kernel 1: transpose+convert Wq,Wk,Wv,Wo (fp32 [k][c]) -> f16 [c][k]
// ---------------------------------------------------------------------------
__global__ void prep_weights_kernel(const float* __restrict__ Wq,
                                    const float* __restrict__ Wk,
                                    const float* __restrict__ Wv,
                                    const float* __restrict__ Wo,
                                    _Float16* __restrict__ wT) {
  const int c = blockIdx.x;          // output column (0..255)
  const int k = threadIdx.x;         // reduction index (0..255)
  const int m = blockIdx.y;          // matrix select
  const float* src = (m == 0) ? Wq : (m == 1) ? Wk : (m == 2) ? Wv : Wo;
  wT[(size_t)m * UNITS * UNITS + (size_t)c * UNITS + k] =
      (_Float16)src[(size_t)k * UNITS + c];
}

// ---------------------------------------------------------------------------
// Kernel 2: fused QKV projection.  X[16384x256] fp32 x WT f16 -> Q,K [b,h,n,d]
// f16 and V transposed [b,h,d,n] f16.  One 16x16 tile per wave, 8 WMMAs.
// ---------------------------------------------------------------------------
__global__ __launch_bounds__(128) void qkv_proj_kernel(
    const float* __restrict__ X, const _Float16* __restrict__ wT,
    _Float16* __restrict__ qh, _Float16* __restrict__ kh,
    _Float16* __restrict__ vt) {
  const int lane = threadIdx.x & 31;
  const int wv   = threadIdx.x >> 5;
  const int l15  = lane & 15;
  const int half = (lane >> 4) & 1;

  const int tileId  = blockIdx.y * 4 + wv;   // 0..47
  const int mat     = tileId >> 4;           // 0=Q 1=K 2=V
  const int colBase = (tileId & 15) * 16;
  const int rowBase = blockIdx.x * 16;
  const int b       = rowBase / NN;
  const int nBase   = rowBase % NN;

  const float*     xrow = X + (size_t)(b * NN + nBase + l15) * UNITS;
  const _Float16*  wrow = wT + (size_t)mat * UNITS * UNITS
                             + (size_t)(colBase + l15) * UNITS;

  f8 acc = {};
  for (int k0 = 0; k0 < UNITS; k0 += 32) {
    // A fragment (fp32 -> f16): runs at k0+half*8 and k0+16+half*8
    const float* pa0 = xrow + k0 + half * 8;
    const float* pa1 = xrow + k0 + 16 + half * 8;
    float4 a0 = ((const float4*)pa0)[0];
    float4 a1 = ((const float4*)pa0)[1];
    float4 a2 = ((const float4*)pa1)[0];
    float4 a3 = ((const float4*)pa1)[1];
    h16 a;
    cvt8(a, 0, a0, a1);
    cvt8(a, 8, a2, a3);
    // B fragment: runs at k0+half*8 and k0+16+half*8 of WT row
    h16 bf = ld_h16(wrow + k0 + half * 8, wrow + k0 + 16 + half * 8);
    acc = __builtin_amdgcn_wmma_f32_16x16x32_f16(false, a, false, bf,
                                                 (short)0, acc, false, false);
  }

#pragma unroll
  for (int r = 0; r < 8; ++r) {
    const int n   = nBase + r + 8 * half;
    const int col = colBase + l15;
    const int hh  = col >> 5;
    const int d   = col & 31;
    const _Float16 val = (_Float16)acc[r];
    if (mat == 0)
      qh[((size_t)(b * NUM_HEADS + hh) * NN + n) * HEAD_DIM + d] = val;
    else if (mat == 1)
      kh[((size_t)(b * NUM_HEADS + hh) * NN + n) * HEAD_DIM + d] = val;
    else
      vt[((size_t)(b * NUM_HEADS + hh) * HEAD_DIM + d) * NN + n] = val;
  }
}

// ---------------------------------------------------------------------------
// Kernel 3: flash attention per (b,h,qtile16).  4 waves split the 2048 keys
// into 32-key chunks; online softmax; LDS merge of per-wave partials.
// Mask comes from the packed bitmask (1 u32 per row covers the whole chunk).
// ---------------------------------------------------------------------------
__global__ __launch_bounds__(128) void flash_attn_kernel(
    const _Float16* __restrict__ qh, const _Float16* __restrict__ kh,
    const _Float16* __restrict__ vt, const unsigned* __restrict__ packed,
    _Float16* __restrict__ ctx16) {
  __shared__ _Float16 stage[4][16][32];     // P tile C->A layout staging
  __shared__ float    ctxPart[4][16][32];   // per-wave ctx partials
  __shared__ float    mPart[4][16];
  __shared__ float    lPart[4][16];

  const int lane = threadIdx.x & 31;
  const int wv   = threadIdx.x >> 5;
  const int l15  = lane & 15;
  const int half = (lane >> 4) & 1;

  const int bh    = blockIdx.y;             // 0..63
  const int b     = bh >> 3;
  const int h     = bh & 7;
  const int qBase = blockIdx.x * 16;

  const _Float16* qp = qh + (size_t)bh * NN * HEAD_DIM;
  const _Float16* kp = kh + (size_t)bh * NN * HEAD_DIM;
  const _Float16* vp = vt + (size_t)bh * HEAD_DIM * NN;
  const unsigned* pq = packed + (size_t)(b * NN + qBase) * NWORDS;

  // Q A-fragment, loaded once per wave
  const _Float16* qrow = qp + (size_t)(qBase + l15) * HEAD_DIM;
  const h16 aq = ld_h16(qrow + half * 16, qrow + half * 16 + 8);

  float m[8], ls[8];
#pragma unroll
  for (int r = 0; r < 8; ++r) { m[r] = -1e30f; ls[r] = 0.f; }
  f8 acc0 = {}, acc1 = {};

  for (int c = wv; c < NN / 32; c += 4) {
    const int kBase = c * 32;

    // prefetch next chunk's K/V rows (gfx1250 global_prefetch_b8)
    if (c + 4 < NN / 32) {
      __builtin_prefetch(kp + (size_t)(kBase + 128 + l15) * HEAD_DIM, 0, 0);
      __builtin_prefetch(vp + (size_t)l15 * NN + kBase + 128, 0, 0);
    }

    // score WMMAs: B = K^T tiles (d rows = half*16+j, k col = l15)
    const _Float16* kr0 = kp + (size_t)(kBase + l15) * HEAD_DIM + half * 16;
    const _Float16* kr1 = kp + (size_t)(kBase + 16 + l15) * HEAD_DIM + half * 16;
    h16 bk0 = ld_h16(kr0, kr0 + 8);
    h16 bk1 = ld_h16(kr1, kr1 + 8);
    f8 z = {};
    f8 s0 = __builtin_amdgcn_wmma_f32_16x16x32_f16(false, aq, false, bk0,
                                                   (short)0, z, false, false);
    f8 s1 = __builtin_amdgcn_wmma_f32_16x16x32_f16(false, aq, false, bk1,
                                                   (short)0, z, false, false);

    // one packed word per row covers all 32 keys of this chunk
    float v0[8], v1[8];
#pragma unroll
    for (int r = 0; r < 8; ++r) {
      const unsigned word = pq[(size_t)(r + 8 * half) * NWORDS + c];
      v0[r] = ((word >> l15) & 1u)        ? s0[r] * SCALE : -1e9f;
      v1[r] = ((word >> (16 + l15)) & 1u) ? s1[r] * SCALE : -1e9f;
    }

#pragma unroll
    for (int r = 0; r < 8; ++r) {
      float cm = fmaxf(v0[r], v1[r]);
      cm = fmaxf(cm, __shfl_xor(cm, 1, 32));
      cm = fmaxf(cm, __shfl_xor(cm, 2, 32));
      cm = fmaxf(cm, __shfl_xor(cm, 4, 32));
      cm = fmaxf(cm, __shfl_xor(cm, 8, 32));    // uniform within 16-lane half
      const float mnew  = fmaxf(m[r], cm);
      const float alpha = __expf(m[r] - mnew);
      m[r] = mnew;
      const float p0 = __expf(v0[r] - mnew);
      const float p1 = __expf(v1[r] - mnew);
      ls[r]   = ls[r] * alpha + p0 + p1;
      acc0[r] *= alpha;
      acc1[r] *= alpha;
      stage[wv][r + 8 * half][l15]      = (_Float16)p0;
      stage[wv][r + 8 * half][16 + l15] = (_Float16)p1;
    }

    // P: C-layout -> A-layout through per-wave LDS stage
    h16 ap = ld_h16(&stage[wv][l15][half * 16], &stage[wv][l15][half * 16 + 8]);

    // V B-fragments (transposed layout -> contiguous over keys)
    const _Float16* vr0 = vp + (size_t)l15 * NN + kBase + half * 16;
    const _Float16* vr1 = vp + (size_t)(16 + l15) * NN + kBase + half * 16;
    h16 bv0 = ld_h16(vr0, vr0 + 8);
    h16 bv1 = ld_h16(vr1, vr1 + 8);
    acc0 = __builtin_amdgcn_wmma_f32_16x16x32_f16(false, ap, false, bv0,
                                                  (short)0, acc0, false, false);
    acc1 = __builtin_amdgcn_wmma_f32_16x16x32_f16(false, ap, false, bv1,
                                                  (short)0, acc1, false, false);
  }

  // reduce row-sums across the 16-lane half; write per-wave partials
#pragma unroll
  for (int r = 0; r < 8; ++r) {
    float s = ls[r];
    s += __shfl_xor(s, 1, 32);
    s += __shfl_xor(s, 2, 32);
    s += __shfl_xor(s, 4, 32);
    s += __shfl_xor(s, 8, 32);
    const int row = r + 8 * half;
    ctxPart[wv][row][l15]      = acc0[r];
    ctxPart[wv][row][16 + l15] = acc1[r];
    if (l15 == 0) { mPart[wv][row] = m[r]; lPart[wv][row] = s; }
  }
  __syncthreads();

  // merge 4 per-wave (m,l,ctx) partials; 512 outputs over 128 threads
  for (int i = threadIdx.x; i < 512; i += 128) {
    const int row = i >> 5;
    const int col = i & 31;
    float M = -1e30f;
#pragma unroll
    for (int w = 0; w < 4; ++w) M = fmaxf(M, mPart[w][row]);
    float sum = 0.f, val = 0.f;
#pragma unroll
    for (int w = 0; w < 4; ++w) {
      const float e = __expf(mPart[w][row] - M);
      sum += lPart[w][row] * e;
      val += ctxPart[w][row][col] * e;
    }
    ctx16[(size_t)(b * NN + qBase + row) * UNITS + h * HEAD_DIM + col] =
        (_Float16)(val / sum);
  }
}

// ---------------------------------------------------------------------------
// Kernel 4: output projection ctx16 [16384x256] x WoT f16 + bo -> fp32 out
// ---------------------------------------------------------------------------
__global__ __launch_bounds__(128) void out_proj_kernel(
    const _Float16* __restrict__ ctx16, const _Float16* __restrict__ woT,
    const float* __restrict__ bo, float* __restrict__ out) {
  const int lane = threadIdx.x & 31;
  const int wv   = threadIdx.x >> 5;
  const int l15  = lane & 15;
  const int half = (lane >> 4) & 1;

  const int colBase = (blockIdx.y * 4 + wv) * 16;
  const int rowBase = blockIdx.x * 16;

  const _Float16* crow = ctx16 + (size_t)(rowBase + l15) * UNITS;
  const _Float16* wrow = woT + (size_t)(colBase + l15) * UNITS;

  f8 acc = {};
  for (int k0 = 0; k0 < UNITS; k0 += 32) {
    h16 a  = ld_h16(crow + k0 + half * 16, crow + k0 + half * 16 + 8);
    h16 bf = ld_h16(wrow + k0 + half * 16, wrow + k0 + half * 16 + 8);
    acc = __builtin_amdgcn_wmma_f32_16x16x32_f16(false, a, false, bf,
                                                 (short)0, acc, false, false);
  }

  const float bias = bo[colBase + l15];
#pragma unroll
  for (int r = 0; r < 8; ++r) {
    out[(size_t)(rowBase + r + 8 * half) * UNITS + colBase + l15] =
        acc[r] + bias;
  }
}

// ---------------------------------------------------------------------------
extern "C" void kernel_launch(void* const* d_in, const int* in_sizes, int n_in,
                              void* d_out, int out_size, void* d_ws,
                              size_t ws_size, hipStream_t stream) {
  const float* X   = (const float*)d_in[0];
  const int*   adj = (const int*)d_in[1];
  const float* Wq  = (const float*)d_in[2];
  const float* Wk  = (const float*)d_in[3];
  const float* Wv  = (const float*)d_in[4];
  const float* Wo  = (const float*)d_in[5];
  const float* bo  = (const float*)d_in[6];
  float* out = (float*)d_out;

  char* ws = (char*)d_ws;
  const size_t WT_HALVES = (size_t)4 * UNITS * UNITS;                // 512 KB
  const size_t T_HALVES  = (size_t)BB * NUM_HEADS * NN * HEAD_DIM;   // 4M halves
  _Float16* wT    = (_Float16*)ws;
  _Float16* qh    = wT + WT_HALVES;
  _Float16* kh    = qh + T_HALVES;
  _Float16* vt    = kh + T_HALVES;
  _Float16* ctx16 = vt + T_HALVES;
  unsigned* packed = (unsigned*)(ctx16 + T_HALVES);  // B*N*64 u32 = 4 MB
  (void)in_sizes; (void)n_in; (void)out_size; (void)ws_size;

  // 0) pack adjacency into bitmask (reads 134 MB once; writes 4 MB)
  pack_adj_kernel<<<dim3(BB * NN), 256, 0, stream>>>(adj, packed);
  // 1) weight transpose/convert
  prep_weights_kernel<<<dim3(UNITS, 4), 256, 0, stream>>>(Wq, Wk, Wv, Wo, wT);
  // 2) fused QKV projection (1024 row-tiles x 48 wave-tiles)
  qkv_proj_kernel<<<dim3(BB * NN / 16, 12), 128, 0, stream>>>(X, wT, qh, kh, vt);
  // 3) flash attention (128 q-tiles x 64 (b,h))
  flash_attn_kernel<<<dim3(NN / 16, BB * NUM_HEADS), 128, 0, stream>>>(
      qh, kh, vt, packed, ctx16);
  // 4) output projection
  out_proj_kernel<<<dim3(BB * NN / 16, 4), 128, 0, stream>>>(
      ctx16, wT + (size_t)3 * UNITS * UNITS, bo, out);
}
